// CapsuleNetwork_23364622090585
// MI455X (gfx1250) — matrix operand order repaired and checked by reading, not compiled, not run on hardware
//
#include <hip/hip_runtime.h>
#include <hip/hip_bf16.h>

// ---------------------------------------------------------------------------
// CDNA5 (gfx1250) CapsuleNetwork forward. All GEMM-like work on
// V_WMMA_F32_16X16X32_BF16. Tiles live in LDS pre-swizzled into the exact
// WMMA VGPR layout (fragment reads = ds_load_b128, staging = b128) with
// ping-pong double buffering (one barrier per K-chunk).
// Block tile: 32M x 64N x 64K, 128 threads (4 waves), 4 WMMA/wave/chunk.
// ---------------------------------------------------------------------------

typedef __attribute__((ext_vector_type(16))) __bf16 v16bf;
typedef __attribute__((ext_vector_type(8)))  float  v8f;

#define EPSF 1e-7f

__device__ __forceinline__ __bf16 f2bf(float x) { return (__bf16)x; }

__device__ __forceinline__ v8f wmma_bf16(v16bf a, v16bf b, v8f c) {
  return __builtin_amdgcn_wmma_f32_16x16x32_bf16(false, a, false, b,
                                                 (short)0, c, false, false);
}

__device__ __forceinline__ float apply_act(float v, int act) {
  if (act == 1) return v > 0.f ? v : 0.f;        // relu
  if (act == 2) return 1.f / (1.f + __expf(-v)); // sigmoid
  return v;
}

// ---- swizzled global formats ----------------------------------------------
// Aswz(M,Kp): [M/32][Kp/64][msub 2][kstep 2][lane 32][t 16]   (2048 el / tile)
//   A frag (16x32 bf16): lane = (m&15) + 16*k[3];  t = (k[4]<<3) | k[2:0]
// Bswz(Kp,N): [Kp/64][N/16][kstep 2][lane 32][t 16]           (1024 el / ktile)
//   B frag (32x16 bf16): lane = (n&15) + 16*k[3];  t = (k[4]<<3) | k[2:0]
__device__ __forceinline__ size_t aswz_idx(int m, int k, int Kp) {
  const int mc = m >> 5, kc = k >> 6;
  const int msub = (m >> 4) & 1, kst = (k >> 5) & 1;
  const int k2 = k & 31;
  const int lane = (m & 15) + (((k2 >> 3) & 1) << 4);
  const int t = ((k2 >> 4) << 3) + (k2 & 7);
  return ((size_t)(mc * (Kp >> 6) + kc) << 11) +
         (((((msub << 1) + kst) << 5) + lane) << 4) + t;
}
__device__ __forceinline__ size_t bswz_idx(int k, int n, int N) {
  const int kc = k >> 6, kst = (k >> 5) & 1;
  const int k2 = k & 31;
  const int lane = (n & 15) + (((k2 >> 3) & 1) << 4);
  return ((size_t)(kc * (N >> 4) + (n >> 4)) << 10) +
         ((((kst << 5) + lane) << 4) + ((k2 >> 4) << 3) + (k2 & 7));
}

// ---------------------------------------------------------------------------
// f32 weights -> bf16 in Bswz layout (pad rows beyond K pre-zeroed by memset)
// ---------------------------------------------------------------------------
__global__ void cvt_swz_b(const float* __restrict__ src, __bf16* __restrict__ dst,
                          int K, int N) {
  long id = (long)blockIdx.x * blockDim.x + threadIdx.x;
  const long total = (long)K * N;
  const long stride = (long)gridDim.x * blockDim.x;
  for (; id < total; id += stride) {
    const int k = (int)(id / N), n = (int)(id % N);
    dst[bswz_idx(k, n, N)] = f2bf(src[id]);
  }
}

// ---------------------------------------------------------------------------
// GEMM: out = act(A @ B + bias).  A: Aswz(M,Kp) bf16, B: Bswz(Kp,N) bf16.
// grid = (M/32, N/64), block = 128. Optionally writes result as Aswz(M,KnP)
// bf16 (feeding the next GEMM) and/or f32 row-major.
// ---------------------------------------------------------------------------
__launch_bounds__(128)
__global__ void gemm_swz_wmma(const __bf16* __restrict__ Aswz,
                              const __bf16* __restrict__ Bswz,
                              const float* __restrict__ bias,
                              float* __restrict__ outF,
                              __bf16* __restrict__ outSwz,
                              int N, int KP, int KnP, int act) {
  __shared__ __align__(32) uint4 AsU[2][256];  // 32x64 bf16, swizzled
  __shared__ __align__(32) uint4 BsU[2][512];  // 64x64 bf16, swizzled
  const int tid = threadIdx.x;
  const int mt = blockIdx.x, nbase = blockIdx.y * 64;
  const int w = tid >> 5, lane = tid & 31, r = lane & 15, h = lane >> 4;
  v8f acc0 = {}, acc1 = {};
  const int kch = KP >> 6;
  const uint4* Ag = (const uint4*)Aswz + (size_t)mt * kch * 256;
  const uint4* Bg = (const uint4*)Bswz + (size_t)(nbase >> 4) * 128;
  const size_t bstride = (size_t)(N >> 4) * 128;

  auto stage = [&](int kc, int sel) {
    const uint4* asrc = Ag + (size_t)kc * 256;
    const uint4* bsrc = Bg + (size_t)kc * bstride;
#pragma unroll
    for (int i = 0; i < 2; ++i) AsU[sel][tid + (i << 7)] = asrc[tid + (i << 7)];
#pragma unroll
    for (int i = 0; i < 4; ++i) BsU[sel][tid + (i << 7)] = bsrc[tid + (i << 7)];
  };

  stage(0, 0);
  __syncthreads();
  for (int kc = 0; kc < kch; ++kc) {
    const int cur = kc & 1;
    if (kc + 1 < kch) stage(kc + 1, cur ^ 1);          // overlap with WMMA
    if (kc + 2 < kch) {
      __builtin_prefetch(Ag + (size_t)(kc + 2) * 256, 0, 3);
      __builtin_prefetch(Bg + (size_t)(kc + 2) * bstride, 0, 3);
    }
    const __bf16* As = (const __bf16*)AsU[cur];
    const __bf16* Bs = (const __bf16*)BsU[cur];
#pragma unroll
    for (int ks = 0; ks < 2; ++ks) {
      const v16bf bf_ = *(const v16bf*)(Bs + (((((w << 1) + ks) << 5) + lane) << 4));
      const v16bf a0  = *(const v16bf*)(As + ((((ks) << 5) + lane) << 4));
      const v16bf a1  = *(const v16bf*)(As + ((((2 + ks) << 5) + lane) << 4));
      acc0 = wmma_bf16(a0, bf_, acc0);
      acc1 = wmma_bf16(a1, bf_, acc1);
    }
    __syncthreads();
  }
  const int n = nbase + (w << 4) + r;
  const float bv = bias ? bias[n] : 0.f;
#pragma unroll
  for (int j = 0; j < 8; ++j) {
    const int m0 = (mt << 5) + j + (h << 3);
    const int m1 = m0 + 16;
    const float v0 = apply_act(acc0[j] + bv, act);
    const float v1 = apply_act(acc1[j] + bv, act);
    if (outF) {
      outF[(size_t)m0 * N + n] = v0;
      outF[(size_t)m1 * N + n] = v1;
    }
    if (outSwz) {
      outSwz[aswz_idx(m0, n, KnP)] = f2bf(v0);
      outSwz[aswz_idx(m1, n, KnP)] = f2bf(v1);
    }
  }
}

// ---------------------------------------------------------------------------
// Implicit-GEMM conv (NHWC, VALID), fully templated geometry so all im2col
// div/mod folds to shifts / mul-by-reciprocal. M = NB*OH*OW (mult of 32),
// N = COUT, K = KS*KS*CIN padded to KP; weights pre-swizzled (Bswz).
// VEC path (bf16 input, CIN%8==0): per-thread fixed (ml,kk8) slot -> one
// global b128 + one ds b128 per chunk, pixel base & LDS dst hoisted.
// ---------------------------------------------------------------------------
template <int IH, int IW, int CIN, int OH, int OW, int COUT,
          int KS, int STR, int KP, bool VEC, int ACT, bool WF, bool WB>
__launch_bounds__(128)
__global__ void conv_swz_wmma(const float* __restrict__ inF,
                              const __bf16* __restrict__ inBf,
                              const __bf16* __restrict__ Bswz,
                              const float* __restrict__ bias,
                              float* __restrict__ outF,
                              __bf16* __restrict__ outB) {
  __shared__ __align__(32) uint4 AsU[2][256];
  __shared__ __align__(32) uint4 BsU[2][512];
  const int tid = threadIdx.x;
  const int mt = blockIdx.x, nbase = blockIdx.y * 64;
  const int w = tid >> 5, lane = tid & 31, r = lane & 15, h = lane >> 4;
  constexpr int K = KS * KS * CIN;
  constexpr int OPIX = OH * OW;
  constexpr int KCH = KP >> 6;
  v8f acc0 = {}, acc1 = {};

  const uint4* Bg = (const uint4*)Bswz + (size_t)(nbase >> 4) * 128;
  constexpr size_t BSTRIDE = (size_t)(COUT >> 4) * 128;

  // hoisted per-thread gather state (VEC path): fixed (ml, kk8) slots
  size_t pixbase[2];
  int adst[2], kk8v[2];
  if (VEC) {
#pragma unroll
    for (int i = 0; i < 2; ++i) {
      const int v = tid + (i << 7);
      const int ml = v >> 3;                 // 0..31
      const int kk8 = (v & 7) << 3;          // 0..56
      const int m = (mt << 5) + ml;
      const int b = m / OPIX, rem = m % OPIX;
      const int oy = rem / OW, ox = rem % OW;
      pixbase[i] = (((size_t)b * IH + oy * STR) * IW + ox * STR) * CIN;
      const int msub = ml >> 4, mm = ml & 15;
      const int kst = kk8 >> 5, base = kk8 & 31;
      const int ll = mm + (((base >> 3) & 1) << 4);
      adst[i] = (((((msub << 1) + kst) << 5) + ll) << 4) + ((base >> 4) << 3);
      kk8v[i] = kk8;
    }
  }

  auto stage = [&](int kc, int sel) {
    const int k0 = kc << 6;
    __bf16* As = (__bf16*)AsU[sel];
    if (VEC) {
#pragma unroll
      for (int i = 0; i < 2; ++i) {
        const int k = k0 + kk8v[i];
        const int ci = k % CIN, kxy = k / CIN;
        const int ky = kxy / KS, kx = kxy % KS;
        *(uint4*)(As + adst[i]) =
            *(const uint4*)(inBf + pixbase[i] + ((size_t)ky * IW + kx) * CIN + ci);
      }
    } else {
#pragma unroll
      for (int i = 0; i < 16; ++i) {         // scalar gather (conv1 only)
        const int e = tid + (i << 7);
        const int ml = e >> 6, kk = e & 63;
        const int k = k0 + kk;
        float v = 0.f;
        if (k < K) {
          const int ci = k % CIN, kxy = k / CIN;
          const int ky = kxy / KS, kx = kxy % KS;
          const int m = (mt << 5) + ml;
          const int b = m / OPIX, rem = m % OPIX;
          const int oy = rem / OW, ox = rem % OW;
          v = inF[(((size_t)b * IH + oy * STR + ky) * IW + ox * STR + kx) * CIN + ci];
        }
        const int msub = ml >> 4, mm = ml & 15;
        const int kst = kk >> 5, k2 = kk & 31;
        const int ll = mm + (((k2 >> 3) & 1) << 4);
        As[(((((msub << 1) + kst) << 5) + ll) << 4) + ((k2 >> 4) << 3) + (k2 & 7)] =
            f2bf(v);
      }
    }
    const uint4* bsrc = Bg + (size_t)kc * BSTRIDE;
#pragma unroll
    for (int i = 0; i < 4; ++i) BsU[sel][tid + (i << 7)] = bsrc[tid + (i << 7)];
  };

  stage(0, 0);
  __syncthreads();
  for (int kc = 0; kc < KCH; ++kc) {
    const int cur = kc & 1;
    if (kc + 1 < KCH) stage(kc + 1, cur ^ 1);
    if (kc + 2 < KCH)
      __builtin_prefetch(Bg + (size_t)(kc + 2) * BSTRIDE, 0, 3);
    const __bf16* As = (const __bf16*)AsU[cur];
    const __bf16* Bs = (const __bf16*)BsU[cur];
#pragma unroll
    for (int ks = 0; ks < 2; ++ks) {
      const v16bf bf_ = *(const v16bf*)(Bs + (((((w << 1) + ks) << 5) + lane) << 4));
      const v16bf a0  = *(const v16bf*)(As + ((((ks) << 5) + lane) << 4));
      const v16bf a1  = *(const v16bf*)(As + ((((2 + ks) << 5) + lane) << 4));
      acc0 = wmma_bf16(a0, bf_, acc0);
      acc1 = wmma_bf16(a1, bf_, acc1);
    }
    __syncthreads();
  }
  const int n = nbase + (w << 4) + r;
  const float bv = bias[n];
#pragma unroll
  for (int j = 0; j < 8; ++j) {
    const int m0 = (mt << 5) + j + (h << 3);
    const int m1 = m0 + 16;
    const float v0 = apply_act(acc0[j] + bv, ACT);
    const float v1 = apply_act(acc1[j] + bv, ACT);
    if (WF) {
      outF[(size_t)m0 * COUT + n] = v0;
      outF[(size_t)m1 * COUT + n] = v1;
    }
    if (WB) {
      outB[(size_t)m0 * COUT + n] = f2bf(v0);
      outB[(size_t)m1 * COUT + n] = f2bf(v1);
    }
  }
}

// ---------------------------------------------------------------------------
// Dynamic routing (u_hat recomputed on the fly; w_pose lives in L2: 47MB<192MB)
// ---------------------------------------------------------------------------
__launch_bounds__(160)
__global__ void routing_accum(const float* __restrict__ u,
                              const float* __restrict__ wpose,
                              const float* __restrict__ logits,
                              float* __restrict__ vbuf,
                              float* __restrict__ vout) {
  __shared__ float sm[160];
  const int b = blockIdx.x;
  const int t = threadIdx.x;          // (cap,d)
  const int cap = t >> 4;
  const float* ub = u + (size_t)b * 9216 * 8;
  const float* lb = logits + (size_t)b * 9216 * 10;
  float s = 0.f;
  for (int n = 0; n < 9216; ++n) {
    float mx = -1e30f;
#pragma unroll
    for (int c2 = 0; c2 < 10; ++c2) mx = fmaxf(mx, lb[n * 10 + c2]);
    float den = 0.f, ecap = 0.f;
#pragma unroll
    for (int c2 = 0; c2 < 10; ++c2) {
      const float e = __expf(lb[n * 10 + c2] - mx);
      den += e;
      if (c2 == cap) ecap = e;
    }
    const float c = ecap / den;
    const float* wp = wpose + (((size_t)n * 10 + cap) * 16 + (t & 15)) * 8;
    const float* up = ub + (size_t)n * 8;
    float uh = 0.f;
#pragma unroll
    for (int p = 0; p < 8; ++p) uh += wp[p] * up[p];
    s += c * uh;
  }
  sm[t] = s;
  __syncthreads();
  float n2 = 0.f;
#pragma unroll
  for (int dd = 0; dd < 16; ++dd) { const float x = sm[cap * 16 + dd]; n2 += x * x; }
  const float nrm = sqrtf(n2);
  const float scale = (n2 / (1.f + n2)) / (nrm + EPSF);
  const float v = scale * s;
  vbuf[(size_t)b * 160 + t] = v;
  if (vout) vout[(size_t)b * 160 + t] = v;
}

__launch_bounds__(160)
__global__ void routing_update(const float* __restrict__ u,
                               const float* __restrict__ wpose,
                               const float* __restrict__ vbuf,
                               float* __restrict__ logits) {
  __shared__ float sm[160];
  const int bn = blockIdx.x;
  const int b = bn / 9216, n = bn % 9216;
  const int t = threadIdx.x, cap = t >> 4, d = t & 15;
  const float* up = u + ((size_t)b * 9216 + n) * 8;
  const float* wp = wpose + (((size_t)n * 10 + cap) * 16 + d) * 8;
  float uh = 0.f;
#pragma unroll
  for (int p = 0; p < 8; ++p) uh += wp[p] * up[p];
  sm[t] = uh * vbuf[(size_t)b * 160 + t];
  __syncthreads();
  if (d == 0) {
    float a = 0.f;
#pragma unroll
    for (int dd = 0; dd < 16; ++dd) a += sm[cap * 16 + dd];
    logits[((size_t)b * 9216 + n) * 10 + cap] += a;
  }
}

// masked v -> decoder input, written directly in Aswz(32,192) layout
__global__ void mask_flatten(const float* __restrict__ vbuf,
                             const float* __restrict__ y,
                             __bf16* __restrict__ vfswz) {
  const int b = blockIdx.x, t = threadIdx.x;  // 160 threads = (cap,d)
  const int cap = t >> 4;
  vfswz[aswz_idx(b, t, 192)] = f2bf(y[b * 10 + cap] * vbuf[(size_t)b * 160 + t]);
}

// ---------------------------------------------------------------------------
extern "C" void kernel_launch(void* const* d_in, const int* in_sizes, int n_in,
                              void* d_out, int out_size, void* d_ws, size_t ws_size,
                              hipStream_t stream) {
  (void)in_sizes; (void)n_in; (void)out_size; (void)ws_size;
  const float* input_x = (const float*)d_in[0];
  const float* y       = (const float*)d_in[1];
  const float* conv1_w = (const float*)d_in[2];
  const float* conv1_b = (const float*)d_in[3];
  const float* conv2_w = (const float*)d_in[4];
  const float* conv2_b = (const float*)d_in[5];
  const float* w_pose  = (const float*)d_in[6];
  const float* w1      = (const float*)d_in[7];
  const float* b1      = (const float*)d_in[8];
  const float* w2      = (const float*)d_in[9];
  const float* b2      = (const float*)d_in[10];
  const float* w3      = (const float*)d_in[11];
  const float* b3      = (const float*)d_in[12];
  float* out = (float*)d_out;          // [0,5120): v ; [5120,...): rec

  char* p = (char*)d_ws;
  auto carve = [&](size_t bytes) -> char* {
    char* q = p; p += (bytes + 255) & ~(size_t)255; return q;
  };
  __bf16* w1c = (__bf16*)carve((size_t)256 * 256 * 2);       // Bswz(256,256)
  __bf16* w2c = (__bf16*)carve((size_t)20736 * 128 * 2);     // Bswz(20736,128)
  __bf16* wm1 = (__bf16*)carve((size_t)192 * 512 * 2);       // Bswz(192,512)
  __bf16* wm2 = (__bf16*)carve((size_t)512 * 1024 * 2);      // Bswz(512,1024)
  __bf16* wm3 = (__bf16*)carve((size_t)1024 * 43200 * 2);    // Bswz(1024,43200)
  __bf16* x1  = (__bf16*)carve((size_t)25690112 * 2);        // conv1 out NHWC
  float*  uB  = (float*) carve((size_t)2359296 * 4);         // u (b,9216,8)
  float*  lg  = (float*) carve((size_t)2949120 * 4);         // logits
  float*  vb  = (float*) carve((size_t)5120 * 4);            // v
  __bf16* vf  = (__bf16*)carve((size_t)32 * 192 * 2);        // Aswz(32,192)
  __bf16* h1  = (__bf16*)carve((size_t)32 * 512 * 2);        // Aswz(32,512)
  __bf16* h2  = (__bf16*)carve((size_t)32 * 1024 * 2);       // Aswz(32,1024)

  // zero the K-padded swizzled buffers, then convert/scatter weights
  hipMemsetAsync(w1c, 0, (size_t)256 * 256 * 2, stream);
  hipMemsetAsync(wm1, 0, (size_t)192 * 512 * 2, stream);
  hipMemsetAsync(vf,  0, (size_t)32 * 192 * 2, stream);
  hipMemsetAsync(lg,  0, (size_t)2949120 * 4, stream);
  cvt_swz_b<<<256,  256, 0, stream>>>(conv1_w, w1c, 243, 256);
  cvt_swz_b<<<2048, 256, 0, stream>>>(conv2_w, w2c, 20736, 128);
  cvt_swz_b<<<256,  256, 0, stream>>>(w1, wm1, 160, 512);
  cvt_swz_b<<<2048, 256, 0, stream>>>(w2, wm2, 512, 1024);
  cvt_swz_b<<<4096, 256, 0, stream>>>(w3, wm3, 1024, 43200);

  // conv1: (32,120,120,3)->relu->(32,56,56,256) bf16.  M=100352 N=256 K=243->256
  conv_swz_wmma<120, 120, 3, 56, 56, 256, 9, 2, 256, false, 1, false, true>
      <<<dim3(3136, 4), 128, 0, stream>>>(input_x, nullptr, w1c, conv1_b,
                                          nullptr, x1);

  // conv2: (32,56,56,256)->(32,24,24,128) f32 == u.  M=18432 N=128 K=20736
  conv_swz_wmma<56, 56, 256, 24, 24, 128, 9, 2, 20736, true, 0, true, false>
      <<<dim3(576, 2), 128, 0, stream>>>(nullptr, x1, w2c, conv2_b, uB, nullptr);

  // dynamic routing, 3 iterations
  for (int it = 0; it < 3; ++it) {
    routing_accum<<<32, 160, 0, stream>>>(uB, w_pose, lg, vb,
                                          it == 2 ? out : nullptr);
    if (it < 2)
      routing_update<<<32 * 9216, 160, 0, stream>>>(uB, w_pose, vb, lg);
  }

  // decoder MLP (M=32; activations chained in swizzled layout)
  mask_flatten<<<32, 160, 0, stream>>>(vb, y, vf);
  gemm_swz_wmma<<<dim3(1, 8),   128, 0, stream>>>(vf, wm1, b1, nullptr, h1,
                                                  512, 192, 512, /*relu*/1);
  gemm_swz_wmma<<<dim3(1, 16),  128, 0, stream>>>(h1, wm2, b2, nullptr, h2,
                                                  1024, 512, 1024, /*relu*/1);
  gemm_swz_wmma<<<dim3(1, 675), 128, 0, stream>>>(h2, wm3, b3, out + 5120,
                                                  nullptr, 43200, 1024, 0,
                                                  /*sigmoid*/2);
}
